// DeformConv_37958920962415
// MI455X (gfx1250) — matrix operand order repaired and checked by reading, not compiled, not run on hardware
//
#include <hip/hip_runtime.h>

// ---------------- problem constants ----------------
#define BN_ 8
#define CIN 256
#define COUT 256
#define HH 56
#define WW 56
#define HW (HH*WW)          // 3136
#define PIX (BN_*HW)        // 25088
#define KTAP 9
#define KDIM (CIN*KTAP)     // 2304
#define KBLK (KDIM/32)      // 72 K-blocks of 32
#define NT (COUT/16)        // 16 N-tiles
#define ARS 264             // A-tile row stride in halfs (256 + 8 pad -> conflict-free frag reads)
#define CRS 260             // C-tile row stride in floats

typedef float v8f  __attribute__((ext_vector_type(8)));
typedef __bf16 v16bf __attribute__((ext_vector_type(16)));

// ---------------- bf16 helpers (round-to-nearest-even) ----------------
__device__ __forceinline__ unsigned short f2bf(float f) {
    union { float f; unsigned int i; } c; c.f = f;
    return (unsigned short)((c.i + 0x7fffu + ((c.i >> 16) & 1u)) >> 16);
}
__device__ __forceinline__ float bflo(unsigned int u) {
    union { unsigned int i; float f; } c; c.i = u << 16; return c.f;
}
__device__ __forceinline__ float bfhi(unsigned int u) {
    union { unsigned int i; float f; } c; c.i = u & 0xffff0000u; return c.f;
}
__device__ __forceinline__ unsigned int packbf2(float a, float b) {
    return (unsigned int)f2bf(a) | ((unsigned int)f2bf(b) << 16);
}
__device__ __forceinline__ unsigned int interp2(unsigned int u00, unsigned int u01,
                                                unsigned int u10, unsigned int u11,
                                                float w00, float w01, float w10, float w11) {
    float a0 = w00*bflo(u00) + w01*bflo(u01) + w10*bflo(u10) + w11*bflo(u11);
    float a1 = w00*bfhi(u00) + w01*bfhi(u01) + w10*bfhi(u10) + w11*bfhi(u11);
    return packbf2(a0, a1);
}

// ---------------- 1) x NCHW f32 -> NHWC bf16 (LDS tiled transpose) ----------------
__global__ void k_transpose(const float* __restrict__ x, unsigned short* __restrict__ xT) {
    __shared__ float tile[32][33];
    const int b   = blockIdx.z;
    const int hw0 = blockIdx.x * 32;
    const int c0  = blockIdx.y * 32;
    const int tx = threadIdx.x, ty = threadIdx.y;   // (32, 8)
#pragma unroll
    for (int j = 0; j < 4; ++j) {
        int c = c0 + ty + 8*j;
        tile[ty + 8*j][tx] = x[((size_t)b*CIN + c)*HW + hw0 + tx]; // coalesced over hw
    }
    __syncthreads();
#pragma unroll
    for (int j = 0; j < 4; ++j) {
        int hw = hw0 + ty + 8*j;
        xT[((size_t)b*HW + hw)*CIN + c0 + tx] = f2bf(tile[tx][ty + 8*j]); // coalesced over c
    }
}

// ---------------- 2) w_conv f32 -> bf16 B-fragment layout ----------------
// layout: wpack[(((kb*16 + nt)*32 + lane)*16 + h], with kappa = kb*32 + (h&7) + (lane>=16 ? 8 : 0) + (h>=8 ? 16 : 0),
// kappa = kk*256 + cin  (kernel-tap major), column n = nt*16 + (lane&15).
__global__ void k_prepack(const float* __restrict__ wconv, unsigned short* __restrict__ wpack) {
    int t = blockIdx.x * 256 + threadIdx.x;    // 72*16*32*16 = 589824
    int h  = t & 15;
    int l  = (t >> 4) & 31;
    int nt = (t >> 9) & 15;
    int kb = t >> 13;
    int n     = nt*16 + (l & 15);
    int kappa = kb*32 + (h & 7) + ((l >= 16) ? 8 : 0) + ((h >= 8) ? 16 : 0);
    int cin = kappa & 255;
    int kk  = kappa >> 8;
    wpack[t] = f2bf(wconv[((size_t)n*CIN + cin)*KTAP + kk]);
}

// ---------------- 3) fold BN params ----------------
__global__ void k_bnparam(const float* __restrict__ gamma, const float* __restrict__ beta,
                          const float* __restrict__ mean, const float* __restrict__ var,
                          float* __restrict__ bn) {
    int o = threadIdx.x;
    float sc = gamma[o] * rsqrtf(var[o] + 1e-5f);
    bn[o] = sc;
    bn[COUT + o] = beta[o] - mean[o] * sc;
}

// ---------------- 4) offset conv, Cin split in 4 chunks (partial sums) ----------------
__global__ void k_off_partial(const float* __restrict__ x, const float* __restrict__ woff,
                              float* __restrict__ part) {
    int t = blockIdx.x * 256 + threadIdx.x;    // 4 * 25088
    int p = t % PIX;
    int chunk = t / PIX;
    int b = p / HW, hw = p % HW;
    int h = hw / WW, w = hw % WW;
    float acc[18];
#pragma unroll
    for (int o = 0; o < 18; ++o) acc[o] = 0.f;
    const float* xb = x + (size_t)b*CIN*HW;
    for (int ci = chunk*64; ci < chunk*64 + 64; ++ci) {
        const float* xc = xb + (size_t)ci*HW;
#pragma unroll
        for (int kh = 0; kh < 3; ++kh) {
            int yy = h - 1 + kh;
            if (yy < 0 || yy >= HH) continue;
#pragma unroll
            for (int kw = 0; kw < 3; ++kw) {
                int xx = w - 1 + kw;
                if (xx < 0 || xx >= WW) continue;
                float xv = xc[yy*WW + xx];              // coalesced (lanes along w)
#pragma unroll
                for (int o = 0; o < 18; ++o)            // weights uniform -> s_loads
                    acc[o] += xv * woff[((o*CIN + ci)*3 + kh)*3 + kw];
            }
        }
    }
    for (int o = 0; o < 18; ++o)
        part[((size_t)(chunk*18 + o))*PIX + p] = acc[o];
}

// ---------------- 5) reduce partials + bias -> off[18][PIX] ----------------
__global__ void k_off_reduce(const float* __restrict__ part, const float* __restrict__ boff,
                             float* __restrict__ off) {
    int i = blockIdx.x * 256 + threadIdx.x;    // 18*25088
    if (i >= 18*PIX) return;
    int oc = i / PIX;
    int p  = i - oc*PIX;
    float s = boff[oc];
#pragma unroll
    for (int c = 0; c < 4; ++c) s += part[((size_t)(c*18 + oc))*PIX + p];
    off[i] = s;
}

// ---------------- 6) fused gather + WMMA GEMM + BN + ReLU ----------------
// Wave mapping: each wave owns 2 M-subtiles x 2 N-subtiles, so each B fragment is
// loaded once and reused for both M rows (halves L2 weight traffic vs 1Mx4N).
union ABFrag { v16bf v; uint4 u[2]; };

__global__ __launch_bounds__(256) void k_main(const unsigned short* __restrict__ xT,
                                              const unsigned short* __restrict__ wpack,
                                              const float* __restrict__ off,
                                              const float* __restrict__ bn,
                                              float* __restrict__ out) {
    __shared__ __align__(16) float smem[32*CRS];     // 33.3 KB; A-tile aliases the front
    unsigned short* At = (unsigned short*)smem;      // [32][ARS] bf16

    const int t   = threadIdx.x;
    const int p0  = blockIdx.x * 32;                 // 32 consecutive pixels (never cross batch)
    const int b   = p0 / HW;
    const int hw0 = p0 % HW;

    const int lane = t & 31;
    const int wv   = t >> 5;                         // 8 waves -> N-subtile pair wv*2, wv*2+1

    // gather decomposition: 8 threads per pixel, 32 channels each
    const int gpix = t >> 3;
    const int gcg  = (t & 7) * 32;
    const int p  = p0 + gpix;
    const int hw = hw0 + gpix;
    const int h = hw / WW, w = hw % WW;

    v8f acc[2][2] = {};                              // [msub][nsub]

    for (int kk = 0; kk < KTAP; ++kk) {
        // ---- gather: bilinear sample 32 pixels x 256 channels into LDS (bf16) ----
        {
            float dy = off[(size_t)(2*kk)   * PIX + p];
            float dx = off[(size_t)(2*kk+1) * PIX + p];
            float py = (float)(h - 1 + kk/3) + dy;
            float px = (float)(w - 1 + kk%3) + dx;
            float fy = floorf(py), fx = floorf(px);
            int y0 = (int)fy, x0 = (int)fx, y1 = y0 + 1, x1 = x0 + 1;
            float ly = py - fy, lx = px - fx;
            float vy0 = (y0 >= 0 && y0 < HH) ? 1.f : 0.f;
            float vy1 = (y1 >= 0 && y1 < HH) ? 1.f : 0.f;
            float vx0 = (x0 >= 0 && x0 < WW) ? 1.f : 0.f;
            float vx1 = (x1 >= 0 && x1 < WW) ? 1.f : 0.f;
            float w00 = (1.f-ly)*(1.f-lx)*vy0*vx0;
            float w01 = (1.f-ly)*lx      *vy0*vx1;
            float w10 = ly*(1.f-lx)      *vy1*vx0;
            float w11 = ly*lx            *vy1*vx1;
            int cy0 = min(max(y0,0),HH-1), cy1 = min(max(y1,0),HH-1);
            int cx0 = min(max(x0,0),WW-1), cx1 = min(max(x1,0),WW-1);
            size_t base = (size_t)b*HW*CIN + gcg;
            const uint4* q00 = (const uint4*)(xT + base + (size_t)(cy0*WW+cx0)*CIN);
            const uint4* q01 = (const uint4*)(xT + base + (size_t)(cy0*WW+cx1)*CIN);
            const uint4* q10 = (const uint4*)(xT + base + (size_t)(cy1*WW+cx0)*CIN);
            const uint4* q11 = (const uint4*)(xT + base + (size_t)(cy1*WW+cx1)*CIN);
            uint4* dst = (uint4*)(At + gpix*ARS + gcg);
#pragma unroll
            for (int i = 0; i < 4; ++i) {            // 8 channels per iter, b128 streams
                uint4 u00 = q00[i], u01 = q01[i], u10 = q10[i], u11 = q11[i];
                uint4 o;
                o.x = interp2(u00.x, u01.x, u10.x, u11.x, w00, w01, w10, w11);
                o.y = interp2(u00.y, u01.y, u10.y, u11.y, w00, w01, w10, w11);
                o.z = interp2(u00.z, u01.z, u10.z, u11.z, w00, w01, w10, w11);
                o.w = interp2(u00.w, u01.w, u10.w, u11.w, w00, w01, w10, w11);
                dst[i] = o;
            }
        }
        __syncthreads();

        // ---- WMMA: 8 K-steps of 32 over this tap's 256 channels ----
        const unsigned short* ar = At + (lane & 15) * ARS;   // msub0 row; msub1 at +16*ARS
        const int hi = (lane >> 4) * 8;                      // A-frag lane-half K offset
        for (int ks = 0; ks < 8; ++ks) {
            const int k0 = ks * 32;
            ABFrag a0, a1;
            a0.u[0] = *(const uint4*)(ar + k0 + hi);                  // K {0..7|8..15}
            a0.u[1] = *(const uint4*)(ar + k0 + 16 + hi);             // K {16..23|24..31}
            a1.u[0] = *(const uint4*)(ar + 16*ARS + k0 + hi);
            a1.u[1] = *(const uint4*)(ar + 16*ARS + k0 + 16 + hi);
            const int kb = kk*8 + ks;
#pragma unroll
            for (int j = 0; j < 2; ++j) {
                const uint4* bp = (const uint4*)(wpack +
                    ((((size_t)kb*NT + (wv*2 + j))*32 + lane) << 4));
                ABFrag bf; bf.u[0] = bp[0]; bf.u[1] = bp[1];
                acc[0][j] = __builtin_amdgcn_wmma_f32_16x16x32_bf16(
                    false, a0.v, false, bf.v, (short)0, acc[0][j], false, false);
                acc[1][j] = __builtin_amdgcn_wmma_f32_16x16x32_bf16(
                    false, a1.v, false, bf.v, (short)0, acc[1][j], false, false);
            }
        }
        __syncthreads();                              // A-tile reused next tap
    }

    // ---- epilogue: C frags -> LDS, then BN+ReLU coalesced store ----
#pragma unroll
    for (int mi = 0; mi < 2; ++mi) {
#pragma unroll
        for (int j = 0; j < 2; ++j) {
            int n = (wv*2 + j)*16 + (lane & 15);
#pragma unroll
            for (int r = 0; r < 8; ++r) {
                int m = mi*16 + r + ((lane >> 4) ? 8 : 0);
                smem[m*CRS + n] = acc[mi][j][r];
            }
        }
    }
    __syncthreads();
    for (int oi = 0; oi < 32; ++oi) {                 // wave handles 32 channels, lanes = pixels
        int o = wv*32 + oi;
        float v = smem[lane*CRS + o] * bn[o] + bn[COUT + o];
        v = v > 0.f ? v : 0.f;
        out[((size_t)(b*COUT + o))*HW + hw0 + lane] = v;  // 128B contiguous per channel
    }
}

// ---------------- workspace layout (bytes) ----------------
//  xT     : 0          .. 12,845,056   (PIX*CIN bf16)
//  wpack  : 12,845,056 .. 14,024,704   (KDIM*COUT bf16)
//  part   : 14,024,704 .. 21,250,048   (4*18*PIX f32)
//  off    : 21,250,048 .. 23,056,384   (18*PIX f32)
//  bn     : 23,056,384 .. 23,058,432   (2*COUT f32)

extern "C" void kernel_launch(void* const* d_in, const int* in_sizes, int n_in,
                              void* d_out, int out_size, void* d_ws, size_t ws_size,
                              hipStream_t stream) {
    const float* x      = (const float*)d_in[0];
    const float* w_off  = (const float*)d_in[1];
    const float* b_off  = (const float*)d_in[2];
    const float* w_conv = (const float*)d_in[3];
    const float* gamma  = (const float*)d_in[4];
    const float* beta   = (const float*)d_in[5];
    const float* rmean  = (const float*)d_in[6];
    const float* rvar   = (const float*)d_in[7];
    float* out = (float*)d_out;

    char* ws = (char*)d_ws;
    unsigned short* xT    = (unsigned short*)(ws);
    unsigned short* wpack = (unsigned short*)(ws + 12845056);
    float* part           = (float*)(ws + 14024704);
    float* offb           = (float*)(ws + 21250048);
    float* bn             = (float*)(ws + 23056384);

    k_transpose  <<<dim3(HW/32, CIN/32, BN_), dim3(32, 8), 0, stream>>>(x, xT);
    k_prepack    <<<(KBLK*NT*32*16)/256, 256, 0, stream>>>(w_conv, wpack);
    k_bnparam    <<<1, 256, 0, stream>>>(gamma, beta, rmean, rvar, bn);
    k_off_partial<<<(4*PIX)/256, 256, 0, stream>>>(x, w_off, part);
    k_off_reduce <<<(18*PIX + 255)/256, 256, 0, stream>>>(part, b_off, offb);
    k_main       <<<PIX/32, 256, 0, stream>>>(xT, wpack, offb, bn, out);
}
